// FeatureInteractionBridge_13365938225639
// MI455X (gfx1250) — matrix operand order repaired and checked by reading, not compiled, not run on hardware
//
#include <hip/hip_runtime.h>
#include <hip/hip_bf16.h>
#include <math.h>
#include <stdint.h>

// ---------------------------------------------------------------------------
// Problem constants (match reference)
// ---------------------------------------------------------------------------
#define BATCH 8
#define SEQL  2048
#define DIM   512
#define BLD   (BATCH * SEQL * DIM)   // 8,388,608 elements

typedef __attribute__((ext_vector_type(16))) __bf16 v16bf;
typedef __attribute__((ext_vector_type(8)))  __bf16 v8bf;
typedef __attribute__((ext_vector_type(8)))  float  v8f;

// ---------------------------------------------------------------------------
// WMMA wrapper: D = A(16x32 bf16) * B(32x16 bf16) + C(16x16 f32)
// ---------------------------------------------------------------------------
__device__ __forceinline__ v8f wmma_bf16(v16bf a, v16bf b, v8f c) {
    return __builtin_amdgcn_wmma_f32_16x16x32_bf16(
        /*neg_a=*/false, a, /*neg_b=*/false, b,
        /*c_mod=*/(short)0, c, /*reuse_a=*/false, /*reuse_b=*/false);
}

// ---------------------------------------------------------------------------
// CDNA5 async global->LDS copy (ASYNCcnt path), GVS addressing:
//   LDS address from the lds_off VGPR; global address = SGPR base + VGPR goff
// ---------------------------------------------------------------------------
__device__ __forceinline__ void async_load_b128(uint32_t lds_off,
                                                const void* base,
                                                uint32_t goff) {
    asm volatile("global_load_async_to_lds_b128 %0, %1, %2"
                 :
                 : "v"(lds_off), "v"(goff), "s"(base)
                 : "memory");
}
__device__ __forceinline__ void wait_asynccnt0() {
    asm volatile("s_wait_asynccnt 0" ::: "memory");
}

// A-fragment (16x32, row-major source, leading dim ld):
//  lane<16 : row = lane,    K chunks [kb+0,kb+8) and [kb+16,kb+24)
//  lane>=16: row = lane-16, K chunks [kb+8,kb+16) and [kb+24,kb+32)
__device__ __forceinline__ v16bf load_a_frag(const __bf16* __restrict__ p,
                                             int ld, int m0, int kb) {
    int lane = threadIdx.x & 31;
    int row  = m0 + (lane & 15);
    int koff = kb + ((lane & 16) ? 8 : 0);
    const __bf16* r = p + (size_t)row * ld + koff;
    v8bf lo = *(const v8bf*)(r);
    v8bf hi = *(const v8bf*)(r + 16);
    v16bf a;
#pragma unroll
    for (int i = 0; i < 8; ++i) { a[i] = lo[i]; a[i + 8] = hi[i]; }
    return a;
}

// B-fragment (32x16): B[k][n] taken from row `n0+lane%16` of a row-major
// matrix whose rows run along K (leading dim ld).
//  lane<16 : col = lane,    K chunks [kb+0,kb+8) and [kb+16,kb+24)
//  lane>=16: col = lane-16, K chunks [kb+8,kb+16) and [kb+24,kb+32)
__device__ __forceinline__ v16bf load_b_frag(const __bf16* __restrict__ p,
                                             int ld, int n0, int kb) {
    int lane = threadIdx.x & 31;
    int col  = n0 + (lane & 15);
    int koff = kb + ((lane & 16) ? 8 : 0);
    const __bf16* r = p + (size_t)col * ld + koff;
    v8bf lo = *(const v8bf*)(r);
    v8bf hi = *(const v8bf*)(r + 16);
    v16bf b;
#pragma unroll
    for (int i = 0; i < 8; ++i) { b[i] = lo[i]; b[i + 8] = hi[i]; }
    return b;
}

// ---------------------------------------------------------------------------
// f32 -> bf16 conversion
// ---------------------------------------------------------------------------
__global__ void cvt_f32_bf16(const float* __restrict__ x,
                             __bf16* __restrict__ y, int n) {
    int i = blockIdx.x * blockDim.x + threadIdx.x;
    if (i < n) y[i] = (__bf16)x[i];
}

// ---------------------------------------------------------------------------
// GEMM: Y[M,N] = X[M,K] * W[N,K]^T + bias[N]   (nn.Linear, K == DIM)
// Output bf16, either row-major (M,N) or per-batch transposed (B, N, Lb).
// Block = 4 waves sharing one 16-row A tile staged in LDS via async copy;
// wave -> one 16x16 tile; grid = (M/16, N/64).
// ---------------------------------------------------------------------------
template <bool TRANS_OUT>
__global__ __launch_bounds__(128)
void gemm_bf16_kernel(const __bf16* __restrict__ X,
                      const __bf16* __restrict__ W,
                      const float*  __restrict__ bias,
                      __bf16* __restrict__ Y,
                      int M, int N, int Lb) {
    constexpr int K = DIM;
    __shared__ __bf16 sA[16 * K];                 // 16 KB A tile

    int tid  = threadIdx.x;
    int wave = tid >> 5;
    int lane = tid & 31;
    int m0 = blockIdx.x * 16;
    int n0 = blockIdx.y * 64 + wave * 16;

    // ---- stage A tile (16 rows x 512 cols bf16) into LDS, async path ----
    // 1024 x 16-byte chunks, 8 per thread, lane-consecutive for coalescing.
    uint32_t ldsBase = (uint32_t)(uintptr_t)(&sA[0]);
#pragma unroll
    for (int j = 0; j < 8; ++j) {
        int c   = j * 128 + tid;                  // chunk id 0..1023
        int row = c >> 5;                         // 32 chunks per row
        int ck  = c & 31;
        uint32_t goff = (uint32_t)((((m0 + row) * K) + ck * 8) * 2);
        async_load_b128(ldsBase + (uint32_t)c * 16u, X, goff);
    }
    wait_asynccnt0();
    __syncthreads();

    // ---- K-loop: A from LDS, B (weights) from global ----
    v8f acc = {};
#pragma unroll 4
    for (int kb = 0; kb < K; kb += 32) {
        v16bf a = load_a_frag((const __bf16*)&sA[0], K, 0, kb);
        v16bf b = load_b_frag(W, K, n0, kb);
        acc = wmma_bf16(a, b, acc);
    }

    int col   = lane & 15;
    int half8 = (lane & 16) ? 8 : 0;
    int n     = n0 + col;
    float bn  = bias[n];
#pragma unroll
    for (int i = 0; i < 8; ++i) {
        int m = m0 + i + half8;
        float v = acc[i] + bn;
        __bf16 h = (__bf16)v;
        if (TRANS_OUT) {
            int bidx = m / Lb;
            int l    = m - bidx * Lb;
            Y[(size_t)bidx * N * Lb + (size_t)n * Lb + l] = h;
        } else {
            Y[(size_t)m * N + n] = h;
        }
    }
}

// ---------------------------------------------------------------------------
// Fused flash-attention with sigmoid gating + residual.
//   Q  : (B,L,D) bf16      Km : (B,L,D) bf16
//   Vt : (B,D,L) bf16      Gl : (B,L,D) bf16 (gate logits)
//   Hq : (B,L,D) f32 residual      Out : (B,L,D) f32
// Block = 4 waves = one 16-query tile; wave w owns features [w*128, w*128+128).
// ---------------------------------------------------------------------------
__global__ __launch_bounds__(128)
void attn_gated_kernel(const __bf16* __restrict__ Q,
                       const __bf16* __restrict__ Km,
                       const __bf16* __restrict__ Vt,
                       const __bf16* __restrict__ Gl,
                       const float*  __restrict__ Hq,
                       float* __restrict__ Out,
                       float scale) {
    constexpr int WAVES = 4;
    constexpr int SLICE = DIM / WAVES;   // 128 features per wave

    int lane  = threadIdx.x & 31;
    int w     = threadIdx.x >> 5;
    int q0    = blockIdx.x * 16;
    int batch = blockIdx.y;

    const __bf16* Qb = Q  + (size_t)batch * SEQL * DIM;
    const __bf16* Kb = Km + (size_t)batch * SEQL * DIM;
    const __bf16* Vb = Vt + (size_t)batch * DIM * SEQL;

    int dBase = w * SLICE;
    int col   = lane & 15;
    int half8 = (lane & 16) ? 8 : 0;

    __shared__ float  sS[WAVES][16 * 32];   // partial scores
    __shared__ __bf16 sP[WAVES][16 * 32];   // probs (A-frag staging)

    // Resident Q fragments for this wave's feature slice
    v16bf qa[4];
#pragma unroll
    for (int ks = 0; ks < 4; ++ks)
        qa[ks] = load_a_frag(Qb, DIM, q0, dBase + ks * 32);

    v8f acc[8];
#pragma unroll
    for (int t = 0; t < 8; ++t) acc[t] = v8f{};
    float mI[8], lI[8];
#pragma unroll
    for (int i = 0; i < 8; ++i) { mI[i] = -3.0e38f; lI[i] = 0.0f; }

    for (int kt = 0; kt < SEQL / 32; ++kt) {
        int kk0 = kt * 32;

        // --- warm next key tile's K and V lines (global_prefetch_b8) ---
        if (kt + 1 < SEQL / 32) {
            int nk = kk0 + 32 + lane;          // 32 key rows across the wave
            __builtin_prefetch(Kb + (size_t)nk * DIM + dBase, 0, 1);
            __builtin_prefetch(Vb + (size_t)(dBase + (lane & 15)) * SEQL
                                  + kk0 + 32, 0, 1);
        }

        // --- partial scores over this wave's 128-feature slice ---
        v8f ps0 = {}, ps1 = {};
#pragma unroll
        for (int ks = 0; ks < 4; ++ks) {
            v16bf b0 = load_b_frag(Kb, DIM, kk0,      dBase + ks * 32);
            v16bf b1 = load_b_frag(Kb, DIM, kk0 + 16, dBase + ks * 32);
            ps0 = wmma_bf16(qa[ks], b0, ps0);
            ps1 = wmma_bf16(qa[ks], b1, ps1);
        }
#pragma unroll
        for (int i = 0; i < 8; ++i) {
            sS[w][(i + half8) * 32 + col]      = ps0[i];
            sS[w][(i + half8) * 32 + 16 + col] = ps1[i];
        }
        __syncthreads();

        // --- cross-wave reduce to full scores (replicated per wave) ---
        float s0[8], s1[8];
#pragma unroll
        for (int i = 0; i < 8; ++i) {
            int r = (i + half8) * 32;
            float a0 = 0.0f, a1 = 0.0f;
#pragma unroll
            for (int ww = 0; ww < WAVES; ++ww) {
                a0 += sS[ww][r + col];
                a1 += sS[ww][r + 16 + col];
            }
            s0[i] = a0 * scale;
            s1[i] = a1 * scale;
        }

        // --- online softmax (row reductions across 16-lane halves) ---
#pragma unroll
        for (int i = 0; i < 8; ++i) {
            float mloc = fmaxf(s0[i], s1[i]);
#pragma unroll
            for (int off = 1; off < 16; off <<= 1)
                mloc = fmaxf(mloc, __shfl_xor(mloc, off, 32));
            float mnew = fmaxf(mI[i], mloc);
            float f    = __expf(mI[i] - mnew);
            float p0   = __expf(s0[i] - mnew);
            float p1   = __expf(s1[i] - mnew);
            float rs   = p0 + p1;
#pragma unroll
            for (int off = 1; off < 16; off <<= 1)
                rs += __shfl_xor(rs, off, 32);
            lI[i] = lI[i] * f + rs;
            mI[i] = mnew;
#pragma unroll
            for (int t = 0; t < 8; ++t) acc[t][i] *= f;
            sP[w][(i + half8) * 32 + col]      = (__bf16)p0;
            sP[w][(i + half8) * 32 + 16 + col] = (__bf16)p1;
        }

        // --- probs -> A-fragment (wave-local LDS round trip) ---
        v16bf pa = load_a_frag((const __bf16*)&sP[w][0], 32, 0, 0);

        // --- context accumulation on this wave's feature slice ---
#pragma unroll
        for (int t = 0; t < 8; ++t) {
            v16bf bv = load_b_frag(Vb, SEQL, dBase + t * 16, kk0);
            acc[t] = wmma_bf16(pa, bv, acc[t]);
        }
        __syncthreads();   // protect sS before next iteration's writes
    }

    // --- epilogue: out = h + sigmoid(gate) * (acc / l) ---
    float inv[8];
#pragma unroll
    for (int i = 0; i < 8; ++i) inv[i] = 1.0f / lI[i];

    size_t base = ((size_t)batch * SEQL + q0) * DIM;
#pragma unroll
    for (int i = 0; i < 8; ++i) {
        size_t rowIdx = base + (size_t)(i + half8) * DIM;
#pragma unroll
        for (int t = 0; t < 8; ++t) {
            int d = dBase + t * 16 + col;
            float ctx = acc[t][i] * inv[i];
            float gl  = (float)Gl[rowIdx + d];
            float g   = 1.0f / (1.0f + __expf(-gl));
            Out[rowIdx + d] = Hq[rowIdx + d] + g * ctx;
        }
    }
}

// ---------------------------------------------------------------------------
// Host-side orchestration
// ---------------------------------------------------------------------------
static inline size_t align256(size_t x) { return (x + 255) & ~(size_t)255; }

extern "C" void kernel_launch(void* const* d_in, const int* in_sizes, int n_in,
                              void* d_out, int out_size, void* d_ws, size_t ws_size,
                              hipStream_t stream) {
    (void)in_sizes; (void)n_in; (void)out_size; (void)ws_size;

    const float* h_s = (const float*)d_in[0];
    const float* h_c = (const float*)d_in[1];
    // names order: qs, kc, vc, qc, ks, vs, gs, gc  (W then b per name)
    const float* Wf[8];
    const float* bf[8];
    for (int i = 0; i < 8; ++i) {
        Wf[i] = (const float*)d_in[2 + 2 * i];
        bf[i] = (const float*)d_in[3 + 2 * i];
    }
    float* out = (float*)d_out;

    // Workspace carving
    char* ws = (char*)d_ws;
    size_t off = 0;
    auto carve = [&](size_t bytes) -> void* {
        void* p = ws + off;
        off += align256(bytes);
        return p;
    };
    __bf16* hs_bf = (__bf16*)carve((size_t)BLD * 2);
    __bf16* hc_bf = (__bf16*)carve((size_t)BLD * 2);
    __bf16* Wbf[8];
    for (int i = 0; i < 8; ++i) Wbf[i] = (__bf16*)carve((size_t)DIM * DIM * 2);
    __bf16* QS  = (__bf16*)carve((size_t)BLD * 2);
    __bf16* KC  = (__bf16*)carve((size_t)BLD * 2);
    __bf16* VCT = (__bf16*)carve((size_t)BLD * 2);
    __bf16* GS  = (__bf16*)carve((size_t)BLD * 2);
    __bf16* QC  = (__bf16*)carve((size_t)BLD * 2);
    __bf16* KS  = (__bf16*)carve((size_t)BLD * 2);
    __bf16* VST = (__bf16*)carve((size_t)BLD * 2);
    __bf16* GC  = (__bf16*)carve((size_t)BLD * 2);

    // 1) conversions
    {
        int n = BLD;
        dim3 g((n + 255) / 256), b(256);
        cvt_f32_bf16<<<g, b, 0, stream>>>(h_s, hs_bf, n);
        cvt_f32_bf16<<<g, b, 0, stream>>>(h_c, hc_bf, n);
        int nw = DIM * DIM;
        dim3 gw((nw + 255) / 256);
        for (int i = 0; i < 8; ++i)
            cvt_f32_bf16<<<gw, b, 0, stream>>>(Wf[i], Wbf[i], nw);
    }

    // 2) projections
    const int M = BATCH * SEQL;
    dim3 gg(M / 16, DIM / 64), gb(128);
    // direction s<-c : q from h_s, k/v from h_c, gate from h_s
    gemm_bf16_kernel<false><<<gg, gb, 0, stream>>>(hs_bf, Wbf[0], bf[0], QS,  M, DIM, SEQL); // Wqs
    gemm_bf16_kernel<false><<<gg, gb, 0, stream>>>(hc_bf, Wbf[1], bf[1], KC,  M, DIM, SEQL); // Wkc
    gemm_bf16_kernel<true ><<<gg, gb, 0, stream>>>(hc_bf, Wbf[2], bf[2], VCT, M, DIM, SEQL); // Wvc (transposed out)
    gemm_bf16_kernel<false><<<gg, gb, 0, stream>>>(hs_bf, Wbf[6], bf[6], GS,  M, DIM, SEQL); // Wgs
    // direction c<-s : q from h_c, k/v from h_s, gate from h_c
    gemm_bf16_kernel<false><<<gg, gb, 0, stream>>>(hc_bf, Wbf[3], bf[3], QC,  M, DIM, SEQL); // Wqc
    gemm_bf16_kernel<false><<<gg, gb, 0, stream>>>(hs_bf, Wbf[4], bf[4], KS,  M, DIM, SEQL); // Wks
    gemm_bf16_kernel<true ><<<gg, gb, 0, stream>>>(hs_bf, Wbf[5], bf[5], VST, M, DIM, SEQL); // Wvs (transposed out)
    gemm_bf16_kernel<false><<<gg, gb, 0, stream>>>(hc_bf, Wbf[7], bf[7], GC,  M, DIM, SEQL); // Wgc

    // 3) fused attention + gate + residual
    float scale = 1.0f / sqrtf((float)DIM);
    dim3 ag(SEQL / 16, BATCH), ab(128);
    attn_gated_kernel<<<ag, ab, 0, stream>>>(QS, KC, VCT, GS, h_s, out,       scale);
    attn_gated_kernel<<<ag, ab, 0, stream>>>(QC, KS, VST, GC, h_c, out + BLD, scale);
}
